// FastSelfAttention_30588757082235
// MI455X (gfx1250) — compile-verified, matrix operand order checked by reading, not run on hardware
//
#include <hip/hip_runtime.h>
#include <hip/hip_bf16.h>

// ---------------------------------------------------------------------------
// FastSelfAttention for MI455X (gfx1250, wave32, WMMA).
// Pipeline:
//   1) cvt f32->bf16 of hidden/context; transpose-convert Wq/Wk/Wt to [N][K]
//   2) gemm_bf16 (WMMA f32_16x16x32_bf16, async global->LDS staging):
//      mixed_q (f32), mixed_k (bf16 + bf16^T)
//   3) scalar score projections -> q_score/k_score (bf16, pre-scaled by 1/8)
//   4) WMMA scores (h=16 padded to K=32) -> [B,Sq,Sk] bf16
//   5) in-place row softmax
//   6) gemm_bf16 batched: context = weights @ mixed_k   (uses mixed_k^T as B)
//   7) gemm_bf16: out = context @ Wt + bt + mixed_q (f32 residual)
// ---------------------------------------------------------------------------

#define BATCH 4
#define SEQ   2048
#define HID   1024
#define NHEAD 16
#define ROWS  (BATCH * SEQ)   // 8192

typedef __attribute__((ext_vector_type(16))) __bf16 v16bf;
typedef __attribute__((ext_vector_type(8)))  float  v8f;

union Frag {
    v16bf v;
    uint4 q[2];
};

__device__ inline float toF(float x) { return x; }
__device__ inline float toF(__hip_bfloat16 x) { return __bfloat162float(x); }

// ---------------------------------------------------------------------------
// CDNA5 async global->LDS copy (16B per lane), tracked by ASYNCcnt.
// LDS operand is the wave-relative LDS byte offset = low 32 bits of the flat
// shared-memory address (ISA 10.2: flat LDS addr maps to LDS by truncation).
// ---------------------------------------------------------------------------
__device__ inline void async_copy16(void* lds, const void* gaddr) {
    unsigned l = (unsigned)(unsigned long long)lds;
    asm volatile("global_load_async_to_lds_b128 %0, %1, off"
                 :: "v"(l), "v"(gaddr) : "memory");
}

__device__ inline void wait_async0() {
    asm volatile("s_wait_asynccnt 0x0" ::: "memory");
}

// ---------------------------------------------------------------------------
// f32 -> bf16 elementwise convert
// ---------------------------------------------------------------------------
__global__ void cvt_f32_bf16(const float* __restrict__ in,
                             __hip_bfloat16* __restrict__ out, long n) {
    long i = (long)blockIdx.x * blockDim.x + threadIdx.x;
    if (i < n) out[i] = __float2bfloat16(in[i]);
}

// ---------------------------------------------------------------------------
// W[K][N] f32 -> WT[N][K] bf16 (32x32 tiles via LDS)
// ---------------------------------------------------------------------------
__global__ __launch_bounds__(256)
void cvt_transpose(const float* __restrict__ W, __hip_bfloat16* __restrict__ WT,
                   int K, int N) {
    __shared__ float t[32][33];
    int k0 = blockIdx.x * 32, n0 = blockIdx.y * 32;
    int tx = threadIdx.x & 31, ty = threadIdx.x >> 5;   // 32 x 8
    for (int i = ty; i < 32; i += 8)
        t[i][tx] = W[(long)(k0 + i) * N + n0 + tx];
    __syncthreads();
    for (int i = ty; i < 32; i += 8)
        WT[(long)(n0 + i) * K + k0 + tx] = __float2bfloat16(t[tx][i]);
}

// ---------------------------------------------------------------------------
// Generic bf16 WMMA GEMM:  C[M,N] = A[M,K] * BT[N,K]^T  (+bias, +residual)
// Block: 256 threads / 8 waves; macro tile 128(M) x 64(N); wave tile 32x32
// (2x2 of v_wmma_f32_16x16x32_bf16); K stepped by 32, double-buffered LDS
// staged with global_load_async_to_lds_b128 (ASYNCcnt) so the copy engine
// overlaps the WMMA stream; prefetch warms L2 two K-tiles ahead.
// Fragment loads follow ISA 16-bit layout: lanes 0-15 -> K 0..7 & 16..23,
// lanes 16-31 -> K 8..15 & 24..31, as two ds_load_b128 each.
// Grid: (M/128, N/64, batches). All dims assumed divisible (true here).
// ---------------------------------------------------------------------------
__global__ __launch_bounds__(256)
void gemm_bf16(const __hip_bfloat16* __restrict__ A, int lda, long aStride,
               const __hip_bfloat16* __restrict__ BT, int ldb, long bStride,
               int K,
               const float* __restrict__ bias,
               const float* __restrict__ residual, long rStride,
               float* __restrict__ outF, long fStride,
               __hip_bfloat16* __restrict__ outB, long oStride,
               __hip_bfloat16* __restrict__ outBT, int ldt, long tStride,
               int ldc) {
    __shared__ __align__(16) __hip_bfloat16 sA[2][128][32];   // 16 KB
    __shared__ __align__(16) __hip_bfloat16 sB[2][64][32];    //  8 KB

    const int tid  = threadIdx.x;
    const int lane = tid & 31;
    const int wave = tid >> 5;
    const int hl   = lane >> 4;        // which 16-lane half
    const int l15  = lane & 15;
    const int wm   = (wave & 3) * 32;  // wave M offset inside macro tile
    const int wn   = (wave >> 2) * 32; // wave N offset inside macro tile
    const long m0  = (long)blockIdx.x * 128;
    const long n0  = (long)blockIdx.y * 64;
    const int  bz  = blockIdx.z;

    const __hip_bfloat16* Ab = A  + (long)bz * aStride;
    const __hip_bfloat16* Bb = BT + (long)bz * bStride;

    // Per-thread staging chunk coordinates (16B chunks, K-tile = 64B/row).
    const int ar0 = tid >> 2;                 // A rows 0..63
    const int ac0 = (tid & 3) * 8;
    const int ar1 = (tid + 256) >> 2;         // A rows 64..127
    const int ac1 = (tid & 3) * 8;
    const int br  = tid >> 2;                 // B rows 0..63
    const int bc  = (tid & 3) * 8;

    v8f acc[2][2] = {};

    auto stage = [&](int kt, int buf) {
        long k0 = (long)kt * 32;
        async_copy16(&sA[buf][ar0][ac0], Ab + (m0 + ar0) * (long)lda + k0 + ac0);
        async_copy16(&sA[buf][ar1][ac1], Ab + (m0 + ar1) * (long)lda + k0 + ac1);
        async_copy16(&sB[buf][br][bc],   Bb + (n0 + br) * (long)ldb + k0 + bc);
    };

    const int nk = K / 32;
    stage(0, 0);
    wait_async0();
    __syncthreads();

    int buf = 0;
    for (int kt = 0; kt < nk; ++kt) {
        if (kt + 1 < nk) stage(kt + 1, buf ^ 1);
        if (kt + 2 < nk) {
            long kp = (long)(kt + 2) * 32;
            __builtin_prefetch(Ab + (m0 + ar0) * (long)lda + kp, 0, 1);
            __builtin_prefetch(Bb + (n0 + br) * (long)ldb + kp, 0, 1);
        }

        Frag fa[2], fb[2];
#pragma unroll
        for (int i = 0; i < 2; ++i) {
            int mr = wm + i * 16 + l15;
            fa[i].q[0] = *(const uint4*)&sA[buf][mr][hl * 8];
            fa[i].q[1] = *(const uint4*)&sA[buf][mr][hl * 8 + 16];
            int nr = wn + i * 16 + l15;
            fb[i].q[0] = *(const uint4*)&sB[buf][nr][hl * 8];
            fb[i].q[1] = *(const uint4*)&sB[buf][nr][hl * 8 + 16];
        }
#pragma unroll
        for (int mi = 0; mi < 2; ++mi)
#pragma unroll
            for (int ni = 0; ni < 2; ++ni)
                acc[mi][ni] = __builtin_amdgcn_wmma_f32_16x16x32_bf16(
                    false, fa[mi].v, false, fb[ni].v,
                    (short)0, acc[mi][ni], false, false);

        wait_async0();      // async writes to buf^1 complete (issued early,
        __syncthreads();    // waited late -> overlapped with WMMA stream)
        buf ^= 1;
    }

    const float* Rb = residual ? residual + (long)bz * rStride : nullptr;
    float* Fb = outF ? outF + (long)bz * fStride : nullptr;
    __hip_bfloat16* Ob = outB ? outB + (long)bz * oStride : nullptr;
    __hip_bfloat16* Tb = outBT ? outBT + (long)bz * tStride : nullptr;

#pragma unroll
    for (int mi = 0; mi < 2; ++mi) {
#pragma unroll
        for (int ni = 0; ni < 2; ++ni) {
            long mb = m0 + wm + mi * 16 + hl * 8;   // first of 8 rows
            long n  = n0 + wn + ni * 16 + l15;
            float bv = bias ? bias[n] : 0.0f;
            float vals[8];
#pragma unroll
            for (int r = 0; r < 8; ++r) {
                float v = acc[mi][ni][r] + bv;
                if (Rb) v += Rb[(mb + r) * (long)ldc + n];
                vals[r] = v;
                if (Fb) Fb[(mb + r) * (long)ldc + n] = v;
                if (Ob) Ob[(mb + r) * (long)ldc + n] = __float2bfloat16(v);
            }
            if (Tb) {
                union { __hip_bfloat16 h[8]; uint4 q; } pk;
#pragma unroll
                for (int r = 0; r < 8; ++r) pk.h[r] = __float2bfloat16(vals[r]);
                *(uint4*)&Tb[n * (long)ldt + mb] = pk.q;   // 8 rows contiguous
            }
        }
    }
}

// ---------------------------------------------------------------------------
// Per-head score projection: score[r][h] = (sum_k mixed[r][k]*Wa[k][h] + ba[h])/8
// Block 256 = 16 rows x 16 heads. Tiny GEMM (0.27 GFLOP), VALU is fine.
// ---------------------------------------------------------------------------
template <typename T>
__global__ __launch_bounds__(256)
void score_kernel(const T* __restrict__ mixed, const float* __restrict__ Wa,
                  const float* __restrict__ ba, __hip_bfloat16* __restrict__ score,
                  int rows) {
    int h = threadIdx.x & 15;
    int r = blockIdx.x * 16 + (threadIdx.x >> 4);
    if (r >= rows) return;
    const T* row = mixed + (long)r * HID;
    float acc = ba[h];
#pragma unroll 8
    for (int k = 0; k < HID; ++k)
        acc += toF(row[k]) * Wa[k * NHEAD + h];
    score[(long)r * NHEAD + h] = __float2bfloat16(acc * 0.125f);
}

// ---------------------------------------------------------------------------
// scores[b][q][k] = sum_h q_score[b,q,h] * k_score[b,k,h]  via WMMA
// h=16 padded to K=32 with a zeroed second fragment chunk.
// Grid (Sq/16, B); 8 waves each own one 16-q tile's A frag and sweep key chunks.
// ---------------------------------------------------------------------------
__global__ __launch_bounds__(256)
void attn_scores_kernel(const __hip_bfloat16* __restrict__ qs,
                        const __hip_bfloat16* __restrict__ ks,
                        __hip_bfloat16* __restrict__ scores) {
    int b    = blockIdx.y;
    int q0   = blockIdx.x * 16;
    int lane = threadIdx.x & 31;
    int wave = threadIdx.x >> 5;
    int hl   = lane >> 4;
    int l15  = lane & 15;
    int koff = hl * 8;

    uint4 zero = make_uint4(0u, 0u, 0u, 0u);
    Frag fa;
    fa.q[0] = *(const uint4*)(qs + ((long)(b * SEQ + q0 + l15)) * NHEAD + koff);
    fa.q[1] = zero;

    for (int c = wave; c < SEQ / 16; c += 8) {
        Frag fb;
        fb.q[0] = *(const uint4*)(ks + ((long)(b * SEQ + c * 16 + l15)) * NHEAD + koff);
        fb.q[1] = zero;
        v8f s = {};
        s = __builtin_amdgcn_wmma_f32_16x16x32_bf16(
            false, fa.v, false, fb.v, (short)0, s, false, false);
        __hip_bfloat16* outp =
            scores + ((long)b * SEQ + q0 + hl * 8) * SEQ + c * 16 + l15;
#pragma unroll
        for (int r = 0; r < 8; ++r)
            outp[(long)r * SEQ] = __float2bfloat16(s[r]);
    }
}

// ---------------------------------------------------------------------------
// In-place row softmax over 2048 bf16 elements. One block per row.
// ---------------------------------------------------------------------------
__global__ __launch_bounds__(256)
void softmax_kernel(__hip_bfloat16* __restrict__ w) {
    long row = blockIdx.x;
    __hip_bfloat16* p = w + row * (long)SEQ;
    int tid = threadIdx.x;
    float v[8];
    float mx = -1e30f;
#pragma unroll
    for (int i = 0; i < 8; ++i) {
        v[i] = __bfloat162float(p[tid + i * 256]);
        mx = fmaxf(mx, v[i]);
    }
    __shared__ float red[256];
    red[tid] = mx;
    __syncthreads();
    for (int s = 128; s > 0; s >>= 1) {
        if (tid < s) red[tid] = fmaxf(red[tid], red[tid + s]);
        __syncthreads();
    }
    mx = red[0];
    __syncthreads();
    float sum = 0.0f;
#pragma unroll
    for (int i = 0; i < 8; ++i) {
        v[i] = __expf(v[i] - mx);
        sum += v[i];
    }
    red[tid] = sum;
    __syncthreads();
    for (int s = 128; s > 0; s >>= 1) {
        if (tid < s) red[tid] += red[tid + s];
        __syncthreads();
    }
    float inv = 1.0f / red[0];
#pragma unroll
    for (int i = 0; i < 8; ++i)
        p[tid + i * 256] = __float2bfloat16(v[i] * inv);
}

// ---------------------------------------------------------------------------
extern "C" void kernel_launch(void* const* d_in, const int* in_sizes, int n_in,
                              void* d_out, int out_size, void* d_ws, size_t ws_size,
                              hipStream_t stream) {
    const float* hs  = (const float*)d_in[0];
    const float* cs  = (const float*)d_in[1];
    const float* Wq  = (const float*)d_in[2];
    const float* bq  = (const float*)d_in[3];
    const float* Wqa = (const float*)d_in[4];
    const float* bqa = (const float*)d_in[5];
    const float* Wk  = (const float*)d_in[6];
    const float* bk  = (const float*)d_in[7];
    const float* Wka = (const float*)d_in[8];
    const float* bka = (const float*)d_in[9];
    const float* Wt  = (const float*)d_in[10];
    const float* bt  = (const float*)d_in[11];
    float* out = (float*)d_out;
    (void)in_sizes; (void)n_in; (void)out_size; (void)ws_size;

    char* w = (char*)d_ws;
    auto alloc = [&](size_t bytes) {
        char* p = w;
        w += (bytes + 255) & ~(size_t)255;
        return p;
    };

    __hip_bfloat16* hsb     = (__hip_bfloat16*)alloc((size_t)ROWS * HID * 2);
    __hip_bfloat16* csb     = (__hip_bfloat16*)alloc((size_t)ROWS * HID * 2);
    __hip_bfloat16* WqT     = (__hip_bfloat16*)alloc((size_t)HID * HID * 2);
    __hip_bfloat16* WkT     = (__hip_bfloat16*)alloc((size_t)HID * HID * 2);
    __hip_bfloat16* WtT     = (__hip_bfloat16*)alloc((size_t)HID * HID * 2);
    float*          mixedq  = (float*)alloc((size_t)ROWS * HID * 4);
    __hip_bfloat16* mixedkb = (__hip_bfloat16*)alloc((size_t)ROWS * HID * 2);
    __hip_bfloat16* mixedkT = (__hip_bfloat16*)alloc((size_t)HID * ROWS * 2); // [HID][ROWS]
    __hip_bfloat16* qsc     = (__hip_bfloat16*)alloc((size_t)ROWS * NHEAD * 2);
    __hip_bfloat16* ksc     = (__hip_bfloat16*)alloc((size_t)ROWS * NHEAD * 2);
    __hip_bfloat16* attw    = (__hip_bfloat16*)alloc((size_t)BATCH * SEQ * SEQ * 2);
    __hip_bfloat16* ctxb    = (__hip_bfloat16*)alloc((size_t)ROWS * HID * 2);

    const long NELT = (long)ROWS * HID;   // 8388608

    // 1) convert activations + transpose-convert weights
    cvt_f32_bf16<<<(int)(NELT / 256), 256, 0, stream>>>(hs, hsb, NELT);
    cvt_f32_bf16<<<(int)(NELT / 256), 256, 0, stream>>>(cs, csb, NELT);
    cvt_transpose<<<dim3(HID / 32, HID / 32), 256, 0, stream>>>(Wq, WqT, HID, HID);
    cvt_transpose<<<dim3(HID / 32, HID / 32), 256, 0, stream>>>(Wk, WkT, HID, HID);
    cvt_transpose<<<dim3(HID / 32, HID / 32), 256, 0, stream>>>(Wt, WtT, HID, HID);

    // 2) projections
    // mixed_q = hs @ Wq + bq  -> f32
    gemm_bf16<<<dim3(ROWS / 128, HID / 64, 1), 256, 0, stream>>>(
        hsb, HID, 0, WqT, HID, 0, HID,
        bq, nullptr, 0, mixedq, 0, nullptr, 0, nullptr, 0, 0, HID);
    // mixed_k = cs @ Wk + bk  -> bf16 (normal) + bf16 transposed [HID][ROWS]
    gemm_bf16<<<dim3(ROWS / 128, HID / 64, 1), 256, 0, stream>>>(
        csb, HID, 0, WkT, HID, 0, HID,
        bk, nullptr, 0, nullptr, 0, mixedkb, 0, mixedkT, ROWS, 0, HID);

    // 3) per-head scores (pre-scaled by 1/sqrt(64))
    score_kernel<float><<<ROWS / 16, 256, 0, stream>>>(mixedq, Wqa, bqa, qsc, ROWS);
    score_kernel<__hip_bfloat16><<<ROWS / 16, 256, 0, stream>>>(mixedkb, Wka, bka, ksc, ROWS);

    // 4) scores = q_score . k_score^T  (WMMA, h padded to 32)
    attn_scores_kernel<<<dim3(SEQ / 16, BATCH), 256, 0, stream>>>(qsc, ksc, attw);

    // 5) softmax in place over key axis
    softmax_kernel<<<ROWS, 256, 0, stream>>>(attw);

    // 6) context = weights @ mixed_k  (batched; B operand = mixed_k^T slice)
    gemm_bf16<<<dim3(SEQ / 128, HID / 64, BATCH), 256, 0, stream>>>(
        attw, SEQ, (long)SEQ * SEQ,
        mixedkT, ROWS, (long)SEQ,            // per-batch column offset b*SEQ
        SEQ,
        nullptr, nullptr, 0, nullptr, 0,
        ctxb, (long)SEQ * HID, nullptr, 0, 0, HID);

    // 7) out = context @ Wt + bt + mixed_q
    gemm_bf16<<<dim3(ROWS / 128, HID / 64, 1), 256, 0, stream>>>(
        ctxb, HID, 0, WtT, HID, 0, HID,
        bt, mixedq, 0, out, 0, nullptr, 0, nullptr, 0, 0, HID);
}
